// PairEmbedding_39195871543953
// MI455X (gfx1250) — compile-verified, matrix operand order checked by reading, not run on hardware
//
#include <hip/hip_runtime.h>
#include <hip/hip_bf16.h>
#include <math.h>

// ---------------- problem constants ----------------
#define NB   4
#define LSEQ 256
#define NA   14          // atoms per residue
#define FEAT 128
#define AA2  196         // NA*NA
#define KD1  224         // 196 padded to 7*32
#define KIN  416         // 410 padded to 13*32
#define MAX_AA 22
#define MAX_RELPOS 32
#define MT   32          // pairs (M rows) per block = 2 WMMA M-tiles

// packed-weight fragment bases (fragment = 32 lanes x 16 bf16 = 512 elems)
#define FB_W1 0          // dist_w1 : 7 ks * 8 nt = 56
#define FB_W2 56         // dist_w2 : 4*8 = 32
#define FB_W3 88         // out_w1  : 13*8 = 104
#define FB_W4 192        // out_w2  : 32
#define FB_W5 224        // out_w3  : 32
#define FRAGS 256        // total -> 256 KB bf16 in workspace
#define SP_OFFSET (FRAGS * 512 * 2)   // byte offset of softplus table in ws
#define SP_N (MAX_AA * MAX_AA * AA2)  // 484*196 softplus table entries

typedef __attribute__((ext_vector_type(16))) __bf16 v16bf;
typedef __attribute__((ext_vector_type(8)))  float  v8f;

// ---------------- helpers ----------------
__device__ __host__ inline unsigned short f2bf_bits(float f) {
  union { float f; unsigned u; } v; v.f = f;
  unsigned r = v.u + 0x7FFFu + ((v.u >> 16) & 1u);
  return (unsigned short)(r >> 16);
}

__device__ inline v8f wmma_bf16(v16bf a, v16bf b, v8f c) {
#if defined(__gfx1250__)
  return __builtin_amdgcn_wmma_f32_16x16x32_bf16(false, a, false, b, (short)0, c,
                                                 false, false);
#else
  (void)a; (void)b; return c;
#endif
}

union FragU { v16bf v; uint4 u[2]; };

// A fragment (16x32 bf16) from LDS, row-major [rows][stride] ushort buffer.
// CDNA5 16-bit A layout: lanes 0-15 row M=l, K {0..7,16..23};
// lanes 16-31 row M=l-16, K {8..15,24..31}.
__device__ inline v16bf load_frag_A(const unsigned short* sbuf, int stride,
                                    int ks, int lane) {
  int row = lane & 15;
  int kb  = ks * 32 + ((lane & 16) ? 8 : 0);
  const unsigned short* p = sbuf + row * stride + kb;
  FragU r;
  r.u[0] = *(const uint4*)(p);        // k_off +0..7
  r.u[1] = *(const uint4*)(p + 16);   // k_off +16..23
  return r.v;
}

// B fragment from packed global weights (fragment-major: frag*512 + lane*16)
__device__ inline v16bf load_frag_B(const unsigned short* wpk, int frag, int lane) {
  const unsigned short* p = wpk + (size_t)frag * 512 + lane * 16;
  FragU r;
  r.u[0] = *(const uint4*)(p);
  r.u[1] = *(const uint4*)(p + 8);
  return r.v;
}

// Two M-tiles (rows 0-15 and 16-31) sharing each B fragment: halves weight
// traffic and interleaves two independent WMMA chains (no D->C RAW stalls).
template <int NKS>
__device__ inline void gemm_acc2(const unsigned short* Abuf, int astride,
                                 const unsigned short* wpk, int fragBase,
                                 int wave, int lane, v8f& c0, v8f& c1) {
  c0 = (v8f){}; c1 = (v8f){};
#pragma unroll
  for (int ks = 0; ks < NKS; ++ks) {
    v16bf b  = load_frag_B(wpk, fragBase + ks * 8 + wave, lane);
    v16bf a0 = load_frag_A(Abuf, astride, ks, lane);
    v16bf a1 = load_frag_A(Abuf + 16 * astride, astride, ks, lane);
    c0 = wmma_bf16(a0, b, c0);
    c1 = wmma_bf16(a1, b, c1);
  }
}

struct F3 { float x, y, z; };
__device__ inline F3 f3sub(F3 a, F3 b) { return {a.x-b.x, a.y-b.y, a.z-b.z}; }
__device__ inline F3 f3cross(F3 a, F3 b) {
  return {a.y*b.z - a.z*b.y, a.z*b.x - a.x*b.z, a.x*b.y - a.y*b.x};
}
__device__ inline float f3dot(F3 a, F3 b) { return a.x*b.x + a.y*b.y + a.z*b.z; }

__device__ inline float dihedral(F3 p0, F3 p1, F3 p2, F3 p3) {
  F3 v0 = f3sub(p2, p1), v1 = f3sub(p0, p1), v2 = f3sub(p3, p2);
  F3 u1 = f3cross(v0, v1), u2 = f3cross(v0, v2);
  float n1 = sqrtf(f3dot(u1, u1)), n2 = sqrtf(f3dot(u2, u2));
  float ca = f3dot(u1, u2) / (n1 * n2);
  ca = fminf(fmaxf(ca, -0.999999f), 0.999999f);
  float s = f3dot(f3cross(v1, v2), v0);
  float sgn = (s > 0.f) ? 1.f : ((s < 0.f) ? -1.f : 0.f);
  float d = sgn * acosf(ca);
  return (d != d) ? 0.f : d;   // nan_to_num
}

__device__ inline F3 load_atom(const float* pos, int b, int l, int a) {
  const float* p = pos + (((size_t)b * LSEQ + l) * NA + a) * 3;
  return {p[0], p[1], p[2]};
}

// ---------------- weight pack kernel (fp32 -> bf16 fragment-major) ----------------
__global__ void prep_weights(const float* __restrict__ w1, const float* __restrict__ w2,
                             const float* __restrict__ w3, const float* __restrict__ w4,
                             const float* __restrict__ w5, unsigned short* __restrict__ out) {
  int idx = blockIdx.x * 256 + threadIdx.x;        // 0 .. FRAGS*512-1
  if (idx >= FRAGS * 512) return;
  int frag = idx >> 9;
  int elem = idx & 511;
  int lane = elem >> 4;
  int t    = elem & 15;

  const float* W; int base, Kact;
  if (frag < FB_W2)      { W = w1; base = FB_W1; Kact = 196; }
  else if (frag < FB_W3) { W = w2; base = FB_W2; Kact = 128; }
  else if (frag < FB_W4) { W = w3; base = FB_W3; Kact = 410; }
  else if (frag < FB_W5) { W = w4; base = FB_W4; Kact = 128; }
  else                   { W = w5; base = FB_W5; Kact = 128; }

  int lf = frag - base;
  int ks = lf >> 3;
  int nt = lf & 7;
  int k_off = (lane < 16) ? ((t < 8) ? t : (t + 8))
                          : ((t < 8) ? (t + 8) : (t + 16));
  int k = ks * 32 + k_off;
  int n = nt * 16 + (lane & 15);
  float v = (k < Kact) ? W[(size_t)k * FEAT + n] : 0.0f;
  out[idx] = f2bf_bits(v);
}

// ---------------- softplus table kernel: sp[aap][ap] = softplus(distcoef) ----------------
__global__ void prep_softplus(const float* __restrict__ distcoef,
                              float* __restrict__ sp) {
  int idx = blockIdx.x * 256 + threadIdx.x;
  if (idx >= SP_N) return;
  sp[idx] = log1pf(expf(distcoef[idx]));
}

// ---------------- fused pair-embedding kernel ----------------
// Block = 256 threads (8 wave32), handles MT=32 pairs (one b,i; 32 consecutive j).
__global__ __launch_bounds__(256)
void pair_embed_kernel(const int* __restrict__ aa, const int* __restrict__ res_nb,
                       const int* __restrict__ chain_nb, const float* __restrict__ pos,
                       const float* __restrict__ mask_atoms,
                       const float* __restrict__ aa_pair_embed,
                       const float* __restrict__ relpos_embed,
                       const float* __restrict__ spTab,
                       const float* __restrict__ dist_b1, const float* __restrict__ dist_b2,
                       const float* __restrict__ out_b1, const float* __restrict__ out_b2,
                       const float* __restrict__ out_b3,
                       const unsigned short* __restrict__ wpk,
                       float* __restrict__ out) {
  __shared__ __align__(16) unsigned short sA[MT * KD1];     // d_gauss (A of GEMM1)
  __shared__ __align__(16) unsigned short sFall[MT * KIN];  // feat_all (A of GEMM3)
  __shared__ __align__(16) unsigned short sH[MT * FEAT];    // hidden buffer (reused)
  __shared__ float sPosI[NA * 3];
  __shared__ float sPosJ[MT * NA * 3];
  __shared__ float sMaskI[NA];
  __shared__ float sMaskJ[MT * NA];
  __shared__ float sDihed[MT * 2];
  __shared__ float sMaskPair[MT];
  __shared__ float sSame[MT];
  __shared__ int   sAAp[MT];
  __shared__ int   sRel[MT];

  const int tid  = threadIdx.x;
  const int lane = tid & 31;
  const int wave = tid >> 5;

  const int bid = blockIdx.x;                 // 0 .. NB*LSEQ*(LSEQ/MT)-1
  const int b   = bid / (LSEQ * (LSEQ / MT));
  const int rem = bid % (LSEQ * (LSEQ / MT));
  const int i   = rem / (LSEQ / MT);
  const int j0  = (rem % (LSEQ / MT)) * MT;

  // hint: warm packed weights (L2 resident across all blocks)
  __builtin_prefetch(wpk + (size_t)wave * 512 * 2, 0, 1);

  // ---- phase 0: residue-pair scalars, positions, dihedrals ----
  if (tid < NA * 3) sPosI[tid] = pos[(((size_t)b * LSEQ + i) * NA) * 3 + tid];
  if (tid < NA)     sMaskI[tid] = mask_atoms[((size_t)b * LSEQ + i) * NA + tid];
  for (int t = tid; t < MT * NA * 3; t += 256) {
    int pj = t / (NA * 3), o = t % (NA * 3);
    sPosJ[t] = pos[(((size_t)b * LSEQ + j0 + pj) * NA) * 3 + o];
  }
  for (int t = tid; t < MT * NA; t += 256) {
    int pj = t / NA;
    sMaskJ[t] = mask_atoms[((size_t)b * LSEQ + j0 + pj) * NA + (t % NA)];
  }
  if (tid < MT) {
    const int pj = tid, j = j0 + pj;
    const int ri = b * LSEQ + i, rj = b * LSEQ + j;
    sAAp[pj] = aa[ri] * MAX_AA + aa[rj];
    sSame[pj] = (chain_nb[ri] == chain_nb[rj]) ? 1.f : 0.f;
    int rel = res_nb[ri] - res_nb[rj];
    rel = rel < -MAX_RELPOS ? -MAX_RELPOS : (rel > MAX_RELPOS ? MAX_RELPOS : rel);
    sRel[pj] = rel;
    sMaskPair[pj] = mask_atoms[(size_t)ri * NA + 1] * mask_atoms[(size_t)rj * NA + 1];
    // dihedrals (N=0, CA=1, C=2)
    F3 pNi = load_atom(pos, b, i, 0), pCAi = load_atom(pos, b, i, 1), pCi = load_atom(pos, b, i, 2);
    F3 pNj = load_atom(pos, b, j, 0), pCAj = load_atom(pos, b, j, 1), pCj = load_atom(pos, b, j, 2);
    sDihed[pj * 2 + 0] = dihedral(pCi, pNj, pCAj, pCj);   // ir_phi
    sDihed[pj * 2 + 1] = dihedral(pNi, pCAi, pCi, pNj);   // ir_psi
  }
  __syncthreads();

  // ---- phase 1: d_gauss = exp(-softplus_tab*d^2)*mask -> sA (bf16, K padded) ----
  for (int t = tid; t < MT * KD1; t += 256) {
    int pj = t / KD1, ap = t % KD1;
    float val = 0.f;
    if (ap < AA2) {
      int a1 = ap / NA, a2 = ap % NA;
      float dx = sPosI[a1 * 3 + 0] - sPosJ[pj * NA * 3 + a2 * 3 + 0];
      float dy = sPosI[a1 * 3 + 1] - sPosJ[pj * NA * 3 + a2 * 3 + 1];
      float dz = sPosI[a1 * 3 + 2] - sPosJ[pj * NA * 3 + a2 * 3 + 2];
      float d2 = (dx * dx + dy * dy + dz * dz) * 0.01f;     // (|d|/10)^2
      float c  = spTab[(size_t)sAAp[pj] * AA2 + ap];        // precomputed softplus
      val = __expf(-c * d2) * sMaskI[a1] * sMaskJ[pj * NA + a2];
    }
    sA[t] = f2bf_bits(val);
  }

  // ---- phase 2a: feat_aapair | feat_relpos -> Fall[:,0:256] ----
  for (int t = tid; t < MT * 256; t += 256) {
    int pj = t >> 8, f = t & 255;
    float v;
    if (f < FEAT) v = aa_pair_embed[(size_t)sAAp[pj] * FEAT + f];
    else          v = relpos_embed[(size_t)(sRel[pj] + MAX_RELPOS) * FEAT + (f - FEAT)] * sSame[pj];
    sFall[pj * KIN + f] = f2bf_bits(v);
  }
  // ---- phase 2b: dihedral angular encodings -> Fall[:,384:416] ----
  {
    const float FREQ[6] = {1.f, 2.f, 3.f, 1.f, 0.5f, 1.f / 3.f};
    for (int t = tid; t < MT * 32; t += 256) {
      int pj = t / 32, f = t % 32;
      float v = 0.f;
      if (f < 26) {
        float ang = sDihed[pj * 2 + f / 13];
        int m = f % 13;
        if (m == 0)      v = ang;
        else if (m <= 6) v = __sinf(ang * FREQ[m - 1]);
        else             v = __cosf(ang * FREQ[m - 7]);
      }
      sFall[pj * KIN + 384 + f] = f2bf_bits(v);
    }
  }
  __syncthreads();

  const int col  = wave * 16 + (lane & 15);   // output column of this lane
  const int rofs = (lane >> 4) * 8;           // C-tile row base (0 / 8)
  v8f c0, c1;

  // ---- GEMM1: relu(d_gauss @ dist_w1 + b1) -> sH ----
  gemm_acc2<7>(sA, KD1, wpk, FB_W1, wave, lane, c0, c1);
  {
    float bb = dist_b1[col];
#pragma unroll
    for (int v = 0; v < 8; ++v) {
      float x0 = c0[v] + bb, x1 = c1[v] + bb;
      sH[(rofs + v) * FEAT + col]      = f2bf_bits(x0 > 0.f ? x0 : 0.f);
      sH[(16 + rofs + v) * FEAT + col] = f2bf_bits(x1 > 0.f ? x1 : 0.f);
    }
  }
  __syncthreads();

  // ---- GEMM2: relu(h @ dist_w2 + b2) -> Fall[:,256:384] (feat_dist) ----
  gemm_acc2<4>(sH, FEAT, wpk, FB_W2, wave, lane, c0, c1);
  {
    float bb = dist_b2[col];
#pragma unroll
    for (int v = 0; v < 8; ++v) {
      float x0 = c0[v] + bb, x1 = c1[v] + bb;
      sFall[(rofs + v) * KIN + 256 + col]      = f2bf_bits(x0 > 0.f ? x0 : 0.f);
      sFall[(16 + rofs + v) * KIN + 256 + col] = f2bf_bits(x1 > 0.f ? x1 : 0.f);
    }
  }
  __syncthreads();

  // ---- GEMM3: relu(feat_all @ out_w1 + b1) -> sH ----
  gemm_acc2<13>(sFall, KIN, wpk, FB_W3, wave, lane, c0, c1);
  {
    float bb = out_b1[col];
#pragma unroll
    for (int v = 0; v < 8; ++v) {
      float x0 = c0[v] + bb, x1 = c1[v] + bb;
      sH[(rofs + v) * FEAT + col]      = f2bf_bits(x0 > 0.f ? x0 : 0.f);
      sH[(16 + rofs + v) * FEAT + col] = f2bf_bits(x1 > 0.f ? x1 : 0.f);
    }
  }
  __syncthreads();

  // ---- GEMM4: relu(h @ out_w2 + b2) -> sH (in place; reads complete first) ----
  gemm_acc2<4>(sH, FEAT, wpk, FB_W4, wave, lane, c0, c1);
  __syncthreads();   // all waves finished reading sH
  {
    float bb = out_b2[col];
#pragma unroll
    for (int v = 0; v < 8; ++v) {
      float x0 = c0[v] + bb, x1 = c1[v] + bb;
      sH[(rofs + v) * FEAT + col]      = f2bf_bits(x0 > 0.f ? x0 : 0.f);
      sH[(16 + rofs + v) * FEAT + col] = f2bf_bits(x1 > 0.f ? x1 : 0.f);
    }
  }
  __syncthreads();

  // ---- GEMM5: (h @ out_w3 + b3) * mask_pair -> global out ----
  gemm_acc2<4>(sH, FEAT, wpk, FB_W5, wave, lane, c0, c1);
  {
    float bb = out_b3[col];
    const size_t rowBase = ((size_t)b * LSEQ + i) * LSEQ + j0;
#pragma unroll
    for (int v = 0; v < 8; ++v) {
      int r0 = rofs + v, r1 = 16 + rofs + v;
      out[(rowBase + r0) * FEAT + col] = (c0[v] + bb) * sMaskPair[r0];
      out[(rowBase + r1) * FEAT + col] = (c1[v] + bb) * sMaskPair[r1];
    }
  }
}

// ---------------- launch ----------------
extern "C" void kernel_launch(void* const* d_in, const int* in_sizes, int n_in,
                              void* d_out, int out_size, void* d_ws, size_t ws_size,
                              hipStream_t stream) {
  (void)in_sizes; (void)n_in; (void)out_size; (void)ws_size;
  const int*   aa            = (const int*)d_in[0];
  const int*   res_nb        = (const int*)d_in[1];
  const int*   chain_nb      = (const int*)d_in[2];
  const float* pos_atoms     = (const float*)d_in[3];
  const float* mask_atoms    = (const float*)d_in[4];
  const float* aa_pair_embed = (const float*)d_in[5];
  const float* relpos_embed  = (const float*)d_in[6];
  const float* distcoef      = (const float*)d_in[7];
  const float* dist_w1       = (const float*)d_in[8];
  const float* dist_b1       = (const float*)d_in[9];
  const float* dist_w2       = (const float*)d_in[10];
  const float* dist_b2       = (const float*)d_in[11];
  const float* out_w1        = (const float*)d_in[12];
  const float* out_b1        = (const float*)d_in[13];
  const float* out_w2        = (const float*)d_in[14];
  const float* out_b2        = (const float*)d_in[15];
  const float* out_w3        = (const float*)d_in[16];
  const float* out_b3        = (const float*)d_in[17];
  unsigned short* wpk = (unsigned short*)d_ws;                 // 256 KB packed bf16 weights
  float* spTab = (float*)((char*)d_ws + SP_OFFSET);            // 371 KB softplus table
  float* out = (float*)d_out;

  prep_weights<<<(FRAGS * 512 + 255) / 256, 256, 0, stream>>>(
      dist_w1, dist_w2, out_w1, out_w2, out_w3, wpk);
  prep_softplus<<<(SP_N + 255) / 256, 256, 0, stream>>>(distcoef, spTab);

  dim3 grid(NB * LSEQ * (LSEQ / MT));   // 8192 blocks, 32 pairs each
  pair_embed_kernel<<<grid, 256, 0, stream>>>(
      aa, res_nb, chain_nb, pos_atoms, mask_atoms, aa_pair_embed, relpos_embed,
      spTab, dist_b1, dist_b2, out_b1, out_b2, out_b3, wpk, out);
}
